// GatedDeltaNetCell_56358560858472
// MI455X (gfx1250) — matrix operand order repaired and checked by reading, not compiled, not run on hardware
//
#include <hip/hip_runtime.h>
#include <hip/hip_bf16.h>
#include <math.h>

// ---------------------------------------------------------------------------
// Gated DeltaNet for MI455X (gfx1250, wave32).
//   Prep   : x -> bf16 (streamed once), weights -> bf16 transposed (N x K),
//            laid out so [WqT|WkT|WbT|WdT|WvT] is one contiguous 384 x D matrix.
//   Stage 1: ONE fused projection GEMM (q,k,beta,delta,v) using
//            v_wmma_f32_16x16x32_bf16 with the shared B-strip staged through
//            LDS via double-buffered global_load_async_to_lds_b128 (ASYNCcnt).
//   Stage 2: sequential gated delta recurrence (VALU + LDS), 1 block/batch.
//   Stage 3: output projection GEMM (same WMMA mainloop) -> fp32 out.
// ---------------------------------------------------------------------------

typedef __attribute__((ext_vector_type(16))) __bf16 v16bf;
typedef __attribute__((ext_vector_type(8)))  __bf16 v8bf;
typedef __attribute__((ext_vector_type(4)))  __bf16 v4bf;
typedef __attribute__((ext_vector_type(8)))  float  v8f;

__device__ __forceinline__ __bf16 f2bf(float f) {
  union { float f; unsigned u; } a; a.f = f;
  unsigned u = a.u;
  u += 0x7FFFu + ((u >> 16) & 1u);          // round-to-nearest-even
  union { unsigned short s; __bf16 b; } r;
  r.s = (unsigned short)(u >> 16);
  return r.b;
}

// ---------------------------------------------------------------------------
// prep kernels
// ---------------------------------------------------------------------------
__global__ __launch_bounds__(256) void gdn_cvt_bf16(
    const float* __restrict__ src, __bf16* __restrict__ dst, size_t n4)
{
  size_t i = (size_t)blockIdx.x * blockDim.x + threadIdx.x;
  const size_t stride = (size_t)gridDim.x * blockDim.x;
  for (; i < n4; i += stride) {
    const float4 f = ((const float4*)src)[i];
    v4bf o; o[0] = f2bf(f.x); o[1] = f2bf(f.y); o[2] = f2bf(f.z); o[3] = f2bf(f.w);
    ((v4bf*)dst)[i] = o;
  }
}

__global__ __launch_bounds__(256) void gdn_transpose_bf16(
    const float* __restrict__ src, __bf16* __restrict__ dst, int R, int C)
{
  int idx = blockIdx.x * blockDim.x + threadIdx.x;
  if (idx < R * C) {
    const int r = idx / C, c = idx % C;
    dst[(size_t)c * R + r] = f2bf(src[idx]);
  }
}

// ---------------------------------------------------------------------------
// shared WMMA mainloop: one wave computes a 16x64 strip (4 accumulators);
// the 64xKd B strip (shared by all 4 waves of the block) is staged through a
// double-buffered LDS tile with async global->LDS copies.
// ---------------------------------------------------------------------------

// stage the 64x32 bf16 chunk Bt[n0..n0+63][k0..k0+31] into BsBuf (4 KB);
// 128 threads x 2 async b128 copies each.
__device__ __forceinline__ void gdn_stage_B(
    const __bf16* __restrict__ Bt, int Kd, int n0, int k0,
    __bf16 (*BsBuf)[32], int tid)
{
  #pragma unroll
  for (int j = 0; j < 2; ++j) {
    const int lin = tid * 2 + j;        // 0..255 sixteen-byte units
    const int row = lin >> 2;           // 0..63
    const int off = (lin & 3) * 8;      // bf16 offset within 64B row
    const __bf16* g = Bt + (size_t)(n0 + row) * Kd + (k0 + off);
    const unsigned lds = (unsigned)(uintptr_t)&BsBuf[row][off];
    asm volatile("global_load_async_to_lds_b128 %0, %1, off"
                 :: "v"(lds), "v"(g) : "memory");
  }
}

__device__ __forceinline__ void gdn_wmma_mainloop(
    const __bf16* __restrict__ A, const __bf16* __restrict__ Bt,
    int Kd, int m0, int n0, int tid, int half, int l16,
    __bf16 (*Bs)[64][32], v8f acc[4])
{
  const __bf16* ar = A + (size_t)(m0 + l16) * Kd;   // this lane's A row

  gdn_stage_B(Bt, Kd, n0, 0, Bs[0], tid);
  for (int k0 = 0, c = 0; k0 < Kd; k0 += 32, ++c) {
    asm volatile("s_wait_asynccnt 0" ::: "memory");  // my staging of chunk c done
    __syncthreads();                                  // everyone's staging done
    if (k0 + 32 < Kd)
      gdn_stage_B(Bt, Kd, n0, k0 + 32, Bs[(c + 1) & 1], tid);  // overlap next

    // A fragment 16x32: lanes 0-15 hold K 0..15, lanes 16-31 hold K 16..31
    union { v16bf v; v8bf h[2]; } a;
    a.h[0] = *(const v8bf*)(ar + k0 + half * 16);
    a.h[1] = *(const v8bf*)(ar + k0 + half * 16 + 8);
    if (k0 + 64 < Kd) __builtin_prefetch(ar + k0 + 64, 0, 1);

    // B fragments from LDS: half 0 holds K 0..15, half 1 holds K 16..31
    __bf16 (*Bc)[32] = Bs[c & 1];
    #pragma unroll
    for (int t = 0; t < 4; ++t) {
      const v16bf b = *(const v16bf*)(&Bc[t * 16 + l16][half * 16]);
      acc[t] = __builtin_amdgcn_wmma_f32_16x16x32_bf16(
          false, a.v, false, b, (short)0, acc[t], false, false);
    }
    // next iteration's barrier protects Bs[c&1] against overwrite
  }
}

// Fused projection GEMM: A[M,D] @ [WqT|WkT|WbT|WdT|WvT]^T  (N = 384)
// group g = n0/64: 0->q 1->k 2->sigmoid(beta)+bb 3->softplus(delta)+bd 4,5->v
__global__ __launch_bounds__(128) void gdn_gemm_proj(
    const __bf16* __restrict__ A, const __bf16* __restrict__ Bt,
    float* __restrict__ Yq, float* __restrict__ Yk, float* __restrict__ Yb,
    float* __restrict__ Yd, float* __restrict__ Yv,
    const float* __restrict__ bb, const float* __restrict__ bd, int Kd)
{
  __shared__ __align__(64) __bf16 Bs[2][64][32];
  const int tid = threadIdx.x;
  const int lane = tid & 31, wave = tid >> 5;
  const int half = lane >> 4, l16 = lane & 15;
  const int m0 = (blockIdx.x * 4 + wave) * 16;
  const int n0 = blockIdx.y * 64;

  v8f acc[4] = {};
  gdn_wmma_mainloop(A, Bt, Kd, m0, n0, tid, half, l16, Bs, acc);

  // block-uniform routing
  const int g = n0 >> 6;
  float* Yt; int ld, act; const float* bias; int nb;
  if (g == 0)      { Yt = Yq; ld = 64;  act = 0; bias = nullptr; nb = n0; }
  else if (g == 1) { Yt = Yk; ld = 64;  act = 0; bias = nullptr; nb = n0 - 64; }
  else if (g == 2) { Yt = Yb; ld = 64;  act = 1; bias = bb;      nb = n0 - 128; }
  else if (g == 3) { Yt = Yd; ld = 64;  act = 2; bias = bd;      nb = n0 - 192; }
  else             { Yt = Yv; ld = 128; act = 0; bias = nullptr; nb = n0 - 256; }

  #pragma unroll
  for (int t = 0; t < 4; ++t) {
    const int n = nb + t * 16 + l16;
    const float bv = bias ? bias[n] : 0.0f;
    #pragma unroll
    for (int r = 0; r < 8; ++r) {
      const int m = m0 + r + 8 * half;              // C/D layout: M = vgpr + 8*half
      float val = acc[t][r] + bv;
      if (act == 1)      val = 1.0f / (1.0f + __expf(-val));
      else if (act == 2) val = (val > 20.0f) ? val : __logf(1.0f + __expf(val));
      Yt[(size_t)m * ld + n] = val;
    }
  }
}

// Plain GEMM (output projection): Y[M,N] = A[M,Kd] @ Bt[N,Kd]^T, fp32 out.
__global__ __launch_bounds__(128) void gdn_gemm_out(
    const __bf16* __restrict__ A, const __bf16* __restrict__ Bt,
    float* __restrict__ Y, int N, int Kd)
{
  __shared__ __align__(64) __bf16 Bs[2][64][32];
  const int tid = threadIdx.x;
  const int lane = tid & 31, wave = tid >> 5;
  const int half = lane >> 4, l16 = lane & 15;
  const int m0 = (blockIdx.x * 4 + wave) * 16;
  const int n0 = blockIdx.y * 64;

  v8f acc[4] = {};
  gdn_wmma_mainloop(A, Bt, Kd, m0, n0, tid, half, l16, Bs, acc);

  #pragma unroll
  for (int t = 0; t < 4; ++t) {
    const int n = n0 + t * 16 + l16;
    #pragma unroll
    for (int r = 0; r < 8; ++r) {
      const int m = m0 + r + 8 * half;
      Y[(size_t)m * N + n] = acc[t][r];
    }
  }
}

// ---------------------------------------------------------------------------
// Sequential scan: S[k,v] = beta_t[k]*S + delta_t[k]*k_t[k]*v_t[v];
// o_t[v] = sum_k q_t[k]*S[k,v].  One block/batch, thread owns 16 state elems.
// ---------------------------------------------------------------------------
__global__ __launch_bounds__(512) void gdn_recurrence(
    const float* __restrict__ q, const float* __restrict__ k,
    const float* __restrict__ v, const float* __restrict__ beta,
    const float* __restrict__ delta, __bf16* __restrict__ o, int T)
{
  const int b   = blockIdx.x;
  const int tid = threadIdx.x;
  const int kr  = tid & 63;             // state row (K)
  const int vb  = (tid >> 6) << 4;      // state col base (V)

  __shared__ float s_q[64], s_k[64], s_b[64], s_d[64], s_v[128];
  __shared__ float part[64][129];       // padded against bank conflicts
  __shared__ float red[512];

  float S[16];
  #pragma unroll
  for (int i = 0; i < 16; ++i) S[i] = 0.0f;

  for (int t = 0; t < T; ++t) {
    const size_t row = (size_t)b * T + t;
    if (tid < 64)       s_q[tid]       = q[row * 64 + tid];
    else if (tid < 128) s_k[tid - 64]  = k[row * 64 + (tid - 64)];
    else if (tid < 192) s_b[tid - 128] = beta[row * 64 + (tid - 128)];
    else if (tid < 256) s_d[tid - 192] = delta[row * 64 + (tid - 192)];
    else if (tid < 384) s_v[tid - 256] = v[row * 128 + (tid - 256)];
    __syncthreads();

    const float qs = s_q[kr];
    const float bs = s_b[kr];
    const float dk = s_d[kr] * s_k[kr];
    #pragma unroll
    for (int i = 0; i < 16; ++i) {
      S[i] = bs * S[i] + dk * s_v[vb + i];
      part[kr][vb + i] = qs * S[i];
    }
    __syncthreads();

    const int vcol = tid >> 2;
    const int c0   = (tid & 3) << 4;
    float r = 0.0f;
    #pragma unroll
    for (int i = 0; i < 16; ++i) r += part[c0 + i][vcol];
    red[tid] = r;
    __syncthreads();

    if (tid < 128) {
      o[row * 128 + tid] = f2bf(
          red[tid * 4] + red[tid * 4 + 1] + red[tid * 4 + 2] + red[tid * 4 + 3]);
    }
    __syncthreads();
  }
}

// ---------------------------------------------------------------------------
extern "C" void kernel_launch(void* const* d_in, const int* in_sizes, int n_in,
                              void* d_out, int out_size, void* d_ws, size_t ws_size,
                              hipStream_t stream) {
  (void)in_sizes; (void)n_in; (void)out_size; (void)ws_size;
  const int B = 8, T = 4096, D = 2048, K = 64, V = 128;
  const int M = B * T;  // 32768 rows

  const float* x  = (const float*)d_in[0];
  const float* Wq = (const float*)d_in[1];
  const float* Wk = (const float*)d_in[2];
  const float* Wv = (const float*)d_in[3];
  const float* Wo = (const float*)d_in[4];
  const float* Wb = (const float*)d_in[5];
  const float* bb = (const float*)d_in[6];
  const float* Wd = (const float*)d_in[7];
  const float* bd = (const float*)d_in[8];
  float* out = (float*)d_out;

  // workspace layout, bf16 buffers first, fp32 after (~187 MB total).
  // wqT..wvT are contiguous and form the fused 384 x D projection matrix.
  __bf16* xb  = (__bf16*)d_ws;                    // M*D
  __bf16* wqT = xb  + (size_t)M * D;              // K*D (fused rows   0..63)
  __bf16* wkT = wqT + (size_t)K * D;              // K*D (fused rows  64..127)
  __bf16* wbT = wkT + (size_t)K * D;              // K*D (fused rows 128..191)
  __bf16* wdT = wbT + (size_t)K * D;              // K*D (fused rows 192..255)
  __bf16* wvT = wdT + (size_t)K * D;              // V*D (fused rows 256..383)
  __bf16* woT = wvT + (size_t)V * D;              // D*V
  __bf16* ob  = woT + (size_t)D * V;              // M*V
  float*  q_buf = (float*)(ob + (size_t)M * V);
  float*  k_buf = q_buf + (size_t)M * K;
  float*  b_buf = k_buf + (size_t)M * K;
  float*  d_buf = b_buf + (size_t)M * K;
  float*  v_buf = d_buf + (size_t)M * K;

  // ---- prep ----
  gdn_cvt_bf16<<<2048, 256, 0, stream>>>(x, xb, (size_t)M * D / 4);
  const int wdk = (D * K + 255) / 256, wdv = (D * V + 255) / 256;
  gdn_transpose_bf16<<<wdk, 256, 0, stream>>>(Wq, wqT, D, K);
  gdn_transpose_bf16<<<wdk, 256, 0, stream>>>(Wk, wkT, D, K);
  gdn_transpose_bf16<<<wdk, 256, 0, stream>>>(Wb, wbT, D, K);
  gdn_transpose_bf16<<<wdk, 256, 0, stream>>>(Wd, wdT, D, K);
  gdn_transpose_bf16<<<wdv, 256, 0, stream>>>(Wv, wvT, D, V);
  gdn_transpose_bf16<<<wdv, 256, 0, stream>>>(Wo, woT, V, D);

  // ---- stage 1: fused projections (N = 384, one pass over x) ----
  dim3 blk(128);
  dim3 gP((M / 16) / 4, 384 / 64);                // (512, 6)
  gdn_gemm_proj<<<gP, blk, 0, stream>>>(xb, wqT, q_buf, k_buf, b_buf, d_buf,
                                        v_buf, bb, bd, D);

  // ---- stage 2: recurrence ----
  gdn_recurrence<<<dim3(B), dim3(512), 0, stream>>>(q_buf, k_buf, v_buf,
                                                    b_buf, d_buf, ob, T);

  // ---- stage 3: output projection (N = 2048, Kd = 128) ----
  dim3 gO((M / 16) / 4, D / 64);                  // (512, 32)
  gdn_gemm_out<<<gO, blk, 0, stream>>>(ob, woT, out, D, V);
}
